// PFTBlock_10385230922120
// MI455X (gfx1250) — matrix operand order repaired and checked by reading, not compiled, not run on hardware
//
#include <hip/hip_runtime.h>
#include <hip/hip_bf16.h>

// ---------------------------------------------------------------------------
// PFT / Swin block for MI455X (gfx1250, wave32, WMMA f32_16x16x32_f16)
// ---------------------------------------------------------------------------

typedef _Float16 half8_t __attribute__((ext_vector_type(8)));
typedef _Float16 v16h    __attribute__((ext_vector_type(16)));
typedef float    v8f     __attribute__((ext_vector_type(8)));

// Problem constants
#define KB    8
#define KH    96
#define KW    96
#define KC    192
#define KHEADS 6
#define KHD   32
#define KN    64          // window tokens (8x8)
#define KNW   144         // windows per image (12x12)
#define KWIN  (KB*KNW)    // 1152 total windows
#define KM    (KB*KH*KW)  // 73728 token rows
#define X_ELEMS ((size_t)KM*KC)          // 14,155,776
#define ATTN_ELEMS ((size_t)KWIN*KHEADS*KN*KN)

// ---------------------------------------------------------------------------
// WMMA fragment loaders (wave32, 16x16x32 f16).
// A (16x32, MxK): lane L (0..31): m = L&15, koff = (L>>4)*16.
//   elements 0..15 = A[m][koff+0..15]  (16 contiguous)
// B (32x16, KxN), stored K-contiguous per column (we keep weights as [N][K]):
//   lane L: n = L&15, kb = (L>>4)*16;  b[i] = Bt[n][kb+i]  (16 contiguous)
// C/D (16x16 f32): lane L holds column n = L&15, rows m = (L>>4)*8 + r.
// ---------------------------------------------------------------------------
__device__ inline v16h load_frag_a(const _Float16* src, int ld, int lane) {
  int m    = lane & 15;
  int koff = (lane >> 4) << 4;
  const _Float16* p = src + (size_t)m * ld + koff;
  return *(const v16h*)(p);
}

__device__ inline v16h load_frag_b(const _Float16* bt, int ld, int lane) {
  int n  = lane & 15;
  int kb = (lane >> 4) << 4;
  const _Float16* p = bt + (size_t)n * ld + kb;
  return *(const v16h*)(p);
}

// ---------------------------------------------------------------------------
// Generic WMMA GEMM: C[M,N] = A[M,K] (f16 row-major) x Bt[N,K] (f16) + bias.
// One wave per 16x64 output tile (4 accumulators): A fragment is loaded once
// per K-step and reused by 4 WMMAs -> 4x less A traffic, load-issue relief.
// Epilogues: EPI=0: store f16 | EPI=1: gelu(tanh)->f16 | EPI=3: +residual->f32
// Requires N % 64 == 0 (true for 576/192/768).
// ---------------------------------------------------------------------------
template <int EPI>
__global__ __launch_bounds__(256) void gemm_wmma(
    const _Float16* __restrict__ A, const _Float16* __restrict__ Bt,
    const float* __restrict__ bias, void* __restrict__ Cout,
    const float* __restrict__ residual, int M, int N, int K) {
  int lane = threadIdx.x & 31;
  int wave = threadIdx.x >> 5;
  int tilesN = N >> 6;                        // 64-wide column tiles
  int totalTiles = (M >> 4) * tilesN;
  int tile = blockIdx.x * (blockDim.x >> 5) + wave;
  if (tile >= totalTiles) return;
  int tm = tile / tilesN, tn4 = tile % tilesN;

  const _Float16* a0 = A  + (size_t)tm * 16 * K;
  const _Float16* b0 = Bt + (size_t)tn4 * 64 * K;

  v8f acc0 = {}, acc1 = {}, acc2 = {}, acc3 = {};
  for (int k0 = 0; k0 < K; k0 += 32) {
    v16h af = load_frag_a(a0 + k0, K, lane);
    // speculative prefetch of A rows two K-steps ahead (global_prefetch_b8)
    __builtin_prefetch(a0 + k0 + 64 + (size_t)(lane & 15) * K, 0, 0);
    v16h bf0 = load_frag_b(b0 + 0 * 16 * (size_t)K + k0, K, lane);
    v16h bf1 = load_frag_b(b0 + 1 * 16 * (size_t)K + k0, K, lane);
    v16h bf2 = load_frag_b(b0 + 2 * 16 * (size_t)K + k0, K, lane);
    v16h bf3 = load_frag_b(b0 + 3 * 16 * (size_t)K + k0, K, lane);
    acc0 = __builtin_amdgcn_wmma_f32_16x16x32_f16(false, af, false, bf0, (short)0, acc0, false, false);
    acc1 = __builtin_amdgcn_wmma_f32_16x16x32_f16(false, af, false, bf1, (short)0, acc1, false, false);
    acc2 = __builtin_amdgcn_wmma_f32_16x16x32_f16(false, af, false, bf2, (short)0, acc2, false, false);
    acc3 = __builtin_amdgcn_wmma_f32_16x16x32_f16(false, af, false, bf3, (short)0, acc3, false, false);
  }

  int mbase = (tm << 4) + ((lane >> 4) << 3);
#pragma unroll
  for (int j = 0; j < 4; ++j) {
    v8f acc = (j == 0) ? acc0 : (j == 1) ? acc1 : (j == 2) ? acc2 : acc3;
    int n = (tn4 << 6) + (j << 4) + (lane & 15);
    float bv = bias[n];
#pragma unroll
    for (int r = 0; r < 8; ++r) {
      float v = acc[r] + bv;
      size_t idx = (size_t)(mbase + r) * N + n;
      if (EPI == 0) {
        ((_Float16*)Cout)[idx] = (_Float16)v;
      } else if (EPI == 1) {
        float g = 0.5f * v * (1.f + tanhf(0.7978845608028654f * (v + 0.044715f * v * v * v)));
        ((_Float16*)Cout)[idx] = (_Float16)g;
      } else {
        ((float*)Cout)[idx] = v + residual[idx];
      }
    }
  }
}

// ---------------------------------------------------------------------------
// Weight transpose + f32->f16 convert: W[K,N] -> Wt[N,K]
// ---------------------------------------------------------------------------
__global__ __launch_bounds__(256) void transpose_w(
    const float* __restrict__ w, _Float16* __restrict__ wt, int K, int N) {
  size_t i = (size_t)blockIdx.x * 256 + threadIdx.x;
  if (i >= (size_t)K * N) return;
  int n = (int)(i % N); int k = (int)(i / N);
  wt[(size_t)n * K + k] = (_Float16)w[i];
}

// ---------------------------------------------------------------------------
// LayerNorm (wave per 192-channel row). windowed=1: also roll(-4,-4) +
// window-partition for LN1; windowed=0: plain flat rows (LN2).
// ---------------------------------------------------------------------------
__global__ __launch_bounds__(256) void layernorm_rows(
    const float* __restrict__ x, const float* __restrict__ g,
    const float* __restrict__ b, _Float16* __restrict__ out, int windowed) {
  int lane = threadIdx.x & 31;
  int wave = threadIdx.x >> 5;
  int row  = blockIdx.x * 8 + wave;     // 0..73727
  if (row >= KM) return;

  const float* xp;
  if (windowed) {
    int wi = row >> 6, n = row & 63;
    int bb = wi / KNW, wl = wi % KNW;
    int wh = wl / 12, ww = wl % 12;
    int hr = wh * 8 + (n >> 3), wr = ww * 8 + (n & 7);
    int hs = (hr + 4) % KH, ws = (wr + 4) % KW;   // roll(-SHIFT)
    xp = x + ((size_t)(bb * KH + hs) * KW + ws) * KC;
  } else {
    xp = x + (size_t)row * KC;
  }

  float v[6];
  float s = 0.f;
#pragma unroll
  for (int i = 0; i < 6; ++i) { v[i] = xp[lane + 32 * i]; s += v[i]; }
#pragma unroll
  for (int o = 16; o; o >>= 1) s += __shfl_xor(s, o, 32);
  float mu = s * (1.f / 192.f);
  float q = 0.f;
#pragma unroll
  for (int i = 0; i < 6; ++i) { float d = v[i] - mu; q += d * d; }
#pragma unroll
  for (int o = 16; o; o >>= 1) q += __shfl_xor(q, o, 32);
  float rstd = rsqrtf(q * (1.f / 192.f) + 1e-5f);

  _Float16* op = out + (size_t)row * KC;
#pragma unroll
  for (int i = 0; i < 6; ++i) {
    int c = lane + 32 * i;
    op[c] = (_Float16)((v[i] - mu) * rstd * g[c] + b[c]);
  }
}

// ---------------------------------------------------------------------------
// Fused shifted-window attention per (window, head): scores via WMMA,
// analytic swin mask, softmax, progressive-focusing renorm with prev_attn
// (written to d_out attn region), attn @ V via WMMA.
// ---------------------------------------------------------------------------
__device__ inline int swin_region(int u) { return u < 88 ? 0 : (u < 92 ? 1 : 2); }

__global__ __launch_bounds__(128) void attn_kernel(
    const _Float16* __restrict__ qkv16, const float* __restrict__ prev_attn,
    float* __restrict__ attn_out, _Float16* __restrict__ attnout16) {
  int w    = blockIdx.x;          // 0..1151
  int h    = blockIdx.y;          // 0..5
  int lane = threadIdx.x & 31;
  int wv   = threadIdx.x >> 5;    // 0..3

  __shared__ alignas(32) float    sc[KN * KN];     // 16 KB
  __shared__ alignas(32) _Float16 at16[KN * KN];   // 8 KB
  __shared__ alignas(32) _Float16 vT[KHD * KN];    // 4 KB

  const _Float16* qbase = qkv16 + (size_t)w * KN * (3 * KC) + h * KHD;
  const _Float16* kbase = qbase + KC;
  const _Float16* vbase = qbase + 2 * KC;

  // transpose V (64x32 -> 32x64) into LDS
  for (int i = threadIdx.x; i < KN * KHD; i += 128) {
    int m = i >> 5, d = i & 31;
    vT[d * KN + m] = vbase[(size_t)m * (3 * KC) + d];
  }

  // scores: wave wv owns query tile row tm = wv (16 queries), 4 key tiles
  {
    int tm = wv;
    v16h a = load_frag_a(qbase + (size_t)tm * 16 * (3 * KC), 3 * KC, lane);
#pragma unroll
    for (int tn = 0; tn < 4; ++tn) {
      v16h b = load_frag_b(kbase + (size_t)tn * 16 * (3 * KC), 3 * KC, lane);
      v8f acc = {};
      acc = __builtin_amdgcn_wmma_f32_16x16x32_f16(
          false, a, false, b, (short)0, acc, false, false);
      int col  = tn * 16 + (lane & 15);
      int row0 = tm * 16 + ((lane >> 4) << 3);
#pragma unroll
      for (int r = 0; r < 8; ++r)
        sc[(row0 + r) * KN + col] = acc[r] * 0.17677669529663687f; // 1/sqrt(32)
    }
  }
  __syncthreads();

  // mask + softmax + progressive focusing (one thread per query row)
  if (threadIdx.x < KN) {
    int nq = threadIdx.x;
    int wl = w % KNW;
    int wh = wl / 12, ww = wl % 12;
    int regn = 3 * swin_region(wh * 8 + (nq >> 3)) + swin_region(ww * 8 + (nq & 7));
    float* row = sc + nq * KN;

    float mx = -1e30f;
    for (int mk = 0; mk < KN; ++mk) {
      int regm = 3 * swin_region(wh * 8 + (mk >> 3)) + swin_region(ww * 8 + (mk & 7));
      float s = row[mk] + (regm != regn ? -100.f : 0.f);
      row[mk] = s;
      mx = fmaxf(mx, s);
    }
    float sum = 0.f;
    for (int mk = 0; mk < KN; ++mk) { float e = __expf(row[mk] - mx); row[mk] = e; sum += e; }
    float inv = 1.f / sum;
    const float* pa = prev_attn + ((size_t)(w * KHEADS + h) * KN + nq) * KN;
    float sum2 = 0.f;
    for (int mk = 0; mk < KN; ++mk) { float a2 = row[mk] * inv * pa[mk]; row[mk] = a2; sum2 += a2; }
    float inv2 = 1.f / (sum2 + 1e-6f);
    float* ao = attn_out + ((size_t)(w * KHEADS + h) * KN + nq) * KN;
    for (int mk = 0; mk < KN; ++mk) {
      float f = row[mk] * inv2;
      ao[mk] = f;
      at16[nq * KN + mk] = (_Float16)f;
    }
  }
  __syncthreads();

  // out = attn(64x64) @ V(64x32): 8 tiles over 4 waves, K=64 -> 2 WMMAs each
  for (int t = wv; t < 8; t += 4) {
    int tm = t >> 1, tn = t & 1;
    v8f acc = {};
#pragma unroll
    for (int k0 = 0; k0 < KN; k0 += 32) {
      v16h a = load_frag_a(at16 + tm * 16 * KN + k0, KN, lane);
      v16h b = load_frag_b(vT + tn * 16 * KN + k0, KN, lane);
      acc = __builtin_amdgcn_wmma_f32_16x16x32_f16(
          false, a, false, b, (short)0, acc, false, false);
    }
    int d    = tn * 16 + (lane & 15);
    int row0 = tm * 16 + ((lane >> 4) << 3);
#pragma unroll
    for (int r = 0; r < 8; ++r)
      attnout16[(size_t)(w * KN + row0 + r) * KC + h * KHD + d] = (_Float16)acc[r];
  }
}

// ---------------------------------------------------------------------------
// LePE: depthwise 3x3 (cross-correlation, SAME, zero pad) over V in window
// layout, added in-place to the attention output (f16).
// ---------------------------------------------------------------------------
__global__ __launch_bounds__(256) void lepe_add(
    const _Float16* __restrict__ qkv16, const float* __restrict__ lk,
    _Float16* __restrict__ ao) {
  size_t i = (size_t)blockIdx.x * 256 + threadIdx.x;   // over KWIN*KN*KC
  if (i >= (size_t)KWIN * KN * KC) return;
  int c = (int)(i % KC); size_t t = i / KC;
  int n = (int)(t % KN); int w = (int)(t / KN);
  int r = n >> 3, cl = n & 7;
  float s = 0.f;
#pragma unroll
  for (int dy = -1; dy <= 1; ++dy)
#pragma unroll
    for (int dx = -1; dx <= 1; ++dx) {
      int rr = r + dy, cc = cl + dx;
      if (rr < 0 || rr > 7 || cc < 0 || cc > 7) continue;
      float vv = (float)qkv16[((size_t)w * KN + rr * 8 + cc) * (3 * KC) + 2 * KC + c];
      s += lk[((dy + 1) * 3 + (dx + 1)) * KC + c] * vv;
    }
  ao[i] = (_Float16)((float)ao[i] + s);
}

// ---------------------------------------------------------------------------
// Window reverse + roll(+4,+4) + residual: x1 = x + scatter(proj)
// ---------------------------------------------------------------------------
__global__ __launch_bounds__(256) void scatter_add(
    const float* __restrict__ x, const _Float16* __restrict__ proj16,
    float* __restrict__ x1) {
  size_t i = (size_t)blockIdx.x * 256 + threadIdx.x;   // over X_ELEMS
  if (i >= X_ELEMS) return;
  int c = (int)(i % KC); size_t sp = i / KC;
  int wp = (int)(sp % KW); size_t t = sp / KW;
  int hp = (int)(t % KH); int bb = (int)(t / KH);
  int hr = (hp + KH - 4) % KH, wr = (wp + KW - 4) % KW;   // un-roll(+SHIFT)
  int wi = bb * KNW + (hr >> 3) * 12 + (wr >> 3);
  int n  = (hr & 7) * 8 + (wr & 7);
  x1[i] = x[i] + (float)proj16[(size_t)(wi * KN + n) * KC + c];
}

// ---------------------------------------------------------------------------
// Launcher
// ---------------------------------------------------------------------------
extern "C" void kernel_launch(void* const* d_in, const int* in_sizes, int n_in,
                              void* d_out, int out_size, void* d_ws, size_t ws_size,
                              hipStream_t stream) {
  const float* x        = (const float*)d_in[0];
  const float* prev     = (const float*)d_in[1];
  const float* n1g      = (const float*)d_in[2];
  const float* n1b      = (const float*)d_in[3];
  const float* n2g      = (const float*)d_in[4];
  const float* n2b      = (const float*)d_in[5];
  const float* qkv_w    = (const float*)d_in[6];
  const float* qkv_b    = (const float*)d_in[7];
  const float* proj_w   = (const float*)d_in[8];
  const float* proj_b   = (const float*)d_in[9];
  const float* lepe_k   = (const float*)d_in[10];
  const float* w1       = (const float*)d_in[11];
  const float* b1       = (const float*)d_in[12];
  const float* w2       = (const float*)d_in[13];
  const float* b2       = (const float*)d_in[14];

  float* out_x    = (float*)d_out;
  float* out_attn = out_x + X_ELEMS;

  // Workspace layout (bytes). R0 is reused: [hwA | qkv16] then mlp1 hidden.
  char* ws = (char*)d_ws;
  const size_t sz_hwA   = (size_t)KM * KC * 2;          // 28,311,552
  const size_t sz_qkv   = (size_t)KM * 3 * KC * 2;      // 84,934,656
  _Float16* hwA      = (_Float16*)(ws);                 // R0 lo
  _Float16* qkv16    = (_Float16*)(ws + sz_hwA);        // R0 hi
  _Float16* mlp1h    = (_Float16*)(ws);                 // R0 (reuse, 113 MB)
  size_t off = sz_hwA + sz_qkv;                          // 113,246,208
  _Float16* attnout16 = (_Float16*)(ws + off); off += sz_hwA;  // R1 (also h2)
  _Float16* h2        = attnout16;                       // reuse after proj
  _Float16* proj16    = (_Float16*)(ws + off); off += sz_hwA;  // R2
  float*    x1        = (float*)(ws + off);    off += (size_t)KM * KC * 4; // R3
  _Float16* qkv_wT  = (_Float16*)(ws + off); off += (size_t)3*KC*KC*2;
  _Float16* proj_wT = (_Float16*)(ws + off); off += (size_t)KC*KC*2;
  _Float16* w1T     = (_Float16*)(ws + off); off += (size_t)4*KC*KC*2;
  _Float16* w2T     = (_Float16*)(ws + off); off += (size_t)4*KC*KC*2;

  // 1) weight transpose/convert
  transpose_w<<<(KC*3*KC + 255)/256, 256, 0, stream>>>(qkv_w,  qkv_wT,  KC,   3*KC);
  transpose_w<<<(KC*KC   + 255)/256, 256, 0, stream>>>(proj_w, proj_wT, KC,   KC);
  transpose_w<<<(KC*4*KC + 255)/256, 256, 0, stream>>>(w1,     w1T,     KC,   4*KC);
  transpose_w<<<(4*KC*KC + 255)/256, 256, 0, stream>>>(w2,     w2T,     4*KC, KC);

  // 2) LN1 + roll + window partition -> hwA (f16)
  layernorm_rows<<<KM/8, 256, 0, stream>>>(x, n1g, n1b, hwA, 1);

  // 3) QKV GEMM: [73728,192] x [192,576]
  {
    int tiles = (KM/16) * (3*KC/64);
    gemm_wmma<0><<<(tiles + 7)/8, 256, 0, stream>>>(hwA, qkv_wT, qkv_b, qkv16,
                                                    nullptr, KM, 3*KC, KC);
  }

  // 4) fused attention (writes attn to d_out + attn@V to attnout16)
  attn_kernel<<<dim3(KWIN, KHEADS), 128, 0, stream>>>(qkv16, prev, out_attn, attnout16);

  // 5) LePE depthwise conv added in place
  lepe_add<<<(int)(((size_t)KWIN*KN*KC + 255)/256), 256, 0, stream>>>(qkv16, lepe_k, attnout16);

  // 6) proj GEMM: [73728,192] x [192,192]
  {
    int tiles = (KM/16) * (KC/64);
    gemm_wmma<0><<<(tiles + 7)/8, 256, 0, stream>>>(attnout16, proj_wT, proj_b, proj16,
                                                    nullptr, KM, KC, KC);
  }

  // 7) window reverse + un-shift + residual -> x1 (f32)
  scatter_add<<<(int)((X_ELEMS + 255)/256), 256, 0, stream>>>(x, proj16, x1);

  // 8) LN2 -> h2 (f16)
  layernorm_rows<<<KM/8, 256, 0, stream>>>(x1, n2g, n2b, h2, 0);

  // 9) MLP1 + gelu: [73728,192] x [192,768] -> f16
  {
    int tiles = (KM/16) * (4*KC/64);
    gemm_wmma<1><<<(tiles + 7)/8, 256, 0, stream>>>(h2, w1T, b1, mlp1h,
                                                    nullptr, KM, 4*KC, KC);
  }

  // 10) MLP2 + residual -> d_out x region (f32)
  {
    int tiles = (KM/16) * (KC/64);
    gemm_wmma<3><<<(tiles + 7)/8, 256, 0, stream>>>(mlp1h, w2T, b2, out_x,
                                                    x1, KM, KC, 4*KC);
  }
  (void)in_sizes; (void)n_in; (void)out_size; (void)ws_size;
}
